// RNN_35785667510977
// MI455X (gfx1250) — matrix operand order repaired and checked by reading, not compiled, not run on hardware
//
#include <hip/hip_runtime.h>
#include <hip/hip_bf16.h>
#include <stdint.h>
#include <math.h>

// Problem constants (match reference)
#define T_STEPS 256
#define BATCH   64
#define DIM     1024
#define HID     1024
#define G4      (4 * HID)
#define EPS_F   0.01f
#define NWG     32           // persistent grid size (HID / 32)

typedef __bf16 bf16_t;
typedef __attribute__((ext_vector_type(16))) __bf16 v16bf;
typedef __attribute__((ext_vector_type(8)))  __bf16 bf16x8;
typedef __attribute__((ext_vector_type(8)))  float  v8f;

// ---------------------------------------------------------------------------
// Fragment loaders (layouts per CDNA5 ISA §7.12.2)
//
// A 16x32 bf16 (MxK): lane holds row m=lane&15.
//   lanes 0-15 : elems 0-7 -> K=k0..k0+7,     elems 8-15 -> K=k0+16..k0+23
//   lanes 16-31: elems 0-7 -> K=k0+8..k0+15,  elems 8-15 -> K=k0+24..k0+31
// => per-lane: two contiguous 8-bf16 (16B) runs at (k0+koffA) and (+16)
//
// B 32x16 bf16 (KxN): lane holds col n=lane&15.
//   same K order as A: lanes 0-15 K=k0..k0+7,k0+16..k0+23; lanes 16-31 K=k0+8..k0+15,k0+24..k0+31
// => per-lane: two 8-bf16 (16B) runs at (k0+koffB) and (+16)
// ---------------------------------------------------------------------------

__device__ inline v16bf frag_B(const bf16_t* p) {
    bf16x8 lo = *(const bf16x8*)(p);
    bf16x8 hi = *(const bf16x8*)(p + 16);
    v16bf r;
#pragma unroll
    for (int i = 0; i < 8; ++i) { r[i] = lo[i]; r[i + 8] = hi[i]; }
    return r;
}

__device__ inline v16bf frag_A(const bf16_t* p) {
    bf16x8 lo = *(const bf16x8*)(p);        // K = base .. base+7
    bf16x8 hi = *(const bf16x8*)(p + 16);   // K = base+16 .. base+23
    v16bf r;
#pragma unroll
    for (int i = 0; i < 8; ++i) { r[i] = lo[i]; r[i + 8] = hi[i]; }
    return r;
}

__device__ inline v8f wmma_bf16(v16bf a, v16bf b, v8f c) {
    // (neg_a, A, neg_b, B, c_mod, C, reuse_a, reuse_b)
    return __builtin_amdgcn_wmma_f32_16x16x32_bf16(false, a, false, b,
                                                   (short)0, c, false, false);
}

__device__ inline float sigmoidf_(float x) {
    return 1.0f / (1.0f + __expf(-x));
}

// Device-wide arrive/release barrier for the persistent grid (NWG blocks).
__device__ inline void grid_barrier(unsigned int* cnt, unsigned int* gen, int t) {
    __syncthreads();
    if (threadIdx.x == 0) {
        __threadfence();                       // release our writes
        unsigned int a = atomicAdd(cnt, 1u);
        if (a == NWG - 1u) {
            atomicExch(cnt, 0u);               // reset for next round
            __threadfence();
            atomicAdd(gen, 1u);                // release generation t+1
        } else {
            while (__hip_atomic_load(gen, __ATOMIC_ACQUIRE,
                                     __HIP_MEMORY_SCOPE_AGENT) <= (unsigned int)t) {
                __builtin_amdgcn_s_sleep(1);
            }
        }
        __threadfence();                       // acquire others' writes
    }
    __syncthreads();
}

// ---------------------------------------------------------------------------
// Prep kernels (run once per kernel_launch; graph-capture safe)
// ---------------------------------------------------------------------------

__global__ void prep_convert(const float* __restrict__ Wih,
                             const float* __restrict__ Whh,
                             const float* __restrict__ bih,
                             const float* __restrict__ bhh,
                             bf16_t* __restrict__ Wbih,
                             bf16_t* __restrict__ Wbhh,
                             float* __restrict__ bsum) {
    size_t i = (size_t)blockIdx.x * 256 + threadIdx.x;   // grid covers 4H*D
    Wbih[i] = (bf16_t)Wih[i];
    Wbhh[i] = (bf16_t)Whh[i];
    if (i < (size_t)G4) bsum[i] = bih[i] + bhh[i];
}

__global__ void prep_zero(uint32_t* __restrict__ p, int n) {
    int i = blockIdx.x * 256 + threadIdx.x;
    if (i < n) p[i] = 0u;
}

__global__ void prep_xnorm(const float* __restrict__ x, float* __restrict__ xnorm) {
    // one block per (t, m) row; fixed-order tree reduction of sum(x^2) over DIM
    const float* xr = x + (size_t)blockIdx.x * DIM;
    float s = 0.0f;
    for (int k = threadIdx.x; k < DIM; k += 256) { float v = xr[k]; s += v * v; }
    __shared__ float red[256];
    red[threadIdx.x] = s;
    __syncthreads();
    for (int off = 128; off > 0; off >>= 1) {
        if (threadIdx.x < off) red[threadIdx.x] += red[threadIdx.x + off];
        __syncthreads();
    }
    if (threadIdx.x == 0) xnorm[blockIdx.x] = red[0];
}

__global__ void prep_x0(const float* __restrict__ x, bf16_t* __restrict__ xb0) {
    int i = blockIdx.x * 256 + threadIdx.x;    // grid covers B*D
    xb0[i] = (bf16_t)x[i];
}

// ---------------------------------------------------------------------------
// Persistent LSTM+MAGE kernel: all T timesteps in one launch.
//   grid = 32 WGs (32 hidden units each), block = 256 (8 waves).
//   wave w -> gate (w>>1), N-tile ((w&1)*16) of this WG's hidden slice.
// ---------------------------------------------------------------------------

__global__ __launch_bounds__(256) void lstm_persist(
    const float*  __restrict__ x,       // [T,B,D] fp32
    const float*  __restrict__ g,       // [T,B,4H]
    const bf16_t* __restrict__ Wbih,    // [4H,D] bf16
    const bf16_t* __restrict__ Wbhh,    // [4H,H] bf16
    const float*  __restrict__ bsum,    // [4H]
    const float*  __restrict__ xnorm,   // [T,B]
    float*        hpart,                // [2][NWG][B]  per-WG |h|^2 partials
    bf16_t*       xb,                   // [2][B,D] bf16 (double buffer)
    bf16_t*       hb,                   // [2][B,H] bf16
    bf16_t*       dhb,                  // [2][B,H] bf16
    float*        c_st,                 // [B,H]
    float*        dc_st,                // [B,H]
    float*        __restrict__ out,     // [2,T,B,H]
    unsigned int* bar_cnt,
    unsigned int* bar_gen)
{
    __shared__ float lds_pre[4 * 64 * 32];   // [gate][m][jl] (plane 0 reused)
    __shared__ float lds_dpre[4 * 64 * 32];

    const int tid   = threadIdx.x;
    const int lane  = tid & 31;
    const int w     = tid >> 5;          // wave 0..7
    const int gate  = w >> 1;
    const int wg    = blockIdx.x;
    const int j0    = wg * 32;           // this WG's hidden-unit slice
    const int jl16  = (w & 1) * 16;
    const int col0  = gate * HID + j0 + jl16;
    const int mlane = lane & 15;
    const int hi    = lane >> 4;
    const int nlane = col0 + mlane;      // this lane's N (weight row)
    const int koffA = hi * 8;
    const int koffB = hi * 8;
    const int jl    = jl16 + mlane;
    const int mrow  = tid >> 5;          // base m for elementwise elements

    const bf16_t* WihRow = Wbih + (size_t)nlane * DIM;
    const bf16_t* WhhRow = Wbhh + (size_t)nlane * HID;
    const float   bv     = bsum[nlane];
    const v8f zacc = {0.f, 0.f, 0.f, 0.f, 0.f, 0.f, 0.f, 0.f};

    for (int t = 0; t < T_STEPS; ++t) {
        const int cur = t & 1;
        const int nxt = (t + 1) & 1;
        const bf16_t* xb_in   = xb  + (size_t)cur * BATCH * DIM;
        const bf16_t* hb_in   = hb  + (size_t)cur * BATCH * HID;
        const bf16_t* dhb_in  = dhb + (size_t)cur * BATCH * HID;
        bf16_t*       hb_out  = hb  + (size_t)nxt * BATCH * HID;
        bf16_t*       dhb_out = dhb + (size_t)nxt * BATCH * HID;
        const float*  hp_in   = hpart + (size_t)cur * NWG * BATCH;
        float*        hp_out  = hpart + (size_t)nxt * NWG * BATCH;

        // --- Phase 1: deterministic |h|^2 per batch row (LDS still free) ---
        if (tid < BATCH) {
            float s = 0.0f;
            for (int q = 0; q < NWG; ++q) s += hp_in[q * BATCH + tid];
            lds_pre[tid] = s;
        }
        __syncthreads();
        float hnr[8];
#pragma unroll
        for (int rr = 0; rr < 8; ++rr) hnr[rr] = lds_pre[mrow + 8 * rr];
        __syncthreads();   // all reads done before GEMM waves touch LDS later

        // --- Phase 2: WMMA GEMMs: pre += x@Wih^T + h@Whh^T ; dpre += dh@Whh^T
        v8f accp[4], accd[4];
#pragma unroll
        for (int mt = 0; mt < 4; ++mt) { accp[mt] = zacc; accd[mt] = zacc; }

        for (int k0 = 0; k0 < DIM; k0 += 32) {
            v16bf bih = frag_B(WihRow + k0 + koffB);
            v16bf bhh = frag_B(WhhRow + k0 + koffB);
#pragma unroll
            for (int mt = 0; mt < 4; ++mt) {
                const int r = mt * 16 + mlane;
                v16bf ax  = frag_A(xb_in  + (size_t)r * DIM + k0 + koffA);
                v16bf ah  = frag_A(hb_in  + (size_t)r * HID + k0 + koffA);
                v16bf adh = frag_A(dhb_in + (size_t)r * HID + k0 + koffA);
                accp[mt] = wmma_bf16(ax,  bih, accp[mt]);
                accp[mt] = wmma_bf16(ah,  bhh, accp[mt]);
                accd[mt] = wmma_bf16(adh, bhh, accd[mt]);
            }
        }

        // --- Phase 3: bias + stage tiles to LDS -----------------------------
        // D layout: VGPR r of a tile holds row m = 16*mt + 8*hi + r, col mlane.
#pragma unroll
        for (int mt = 0; mt < 4; ++mt) {
#pragma unroll
            for (int r2 = 0; r2 < 8; ++r2) {
                const int m = mt * 16 + hi * 8 + r2;
                lds_pre [gate * 2048 + m * 32 + jl] = accp[mt][r2] + bv;
                lds_dpre[gate * 2048 + m * 32 + jl] = accd[mt][r2];
            }
        }
        __syncthreads();

        // --- Phase 4: elementwise LSTM + MAGE tangents (8 elems/thread) ----
        const float* gbase = g + (size_t)t * BATCH * G4;
#pragma unroll
        for (int rr = 0; rr < 8; ++rr) {
            const int idx = tid + rr * 256;
            const int m   = idx >> 5;
            const int jj  = idx & 31;
            const int j   = j0 + jj;

            const float pi = lds_pre[0 * 2048 + m * 32 + jj];
            const float pf = lds_pre[1 * 2048 + m * 32 + jj];
            const float pg = lds_pre[2 * 2048 + m * 32 + jj];
            const float po = lds_pre[3 * 2048 + m * 32 + jj];
            const float dhi = lds_dpre[0 * 2048 + m * 32 + jj];
            const float dhf = lds_dpre[1 * 2048 + m * 32 + jj];
            const float dhg = lds_dpre[2 * 2048 + m * 32 + jj];
            const float dho = lds_dpre[3 * 2048 + m * 32 + jj];

            const float* gr = gbase + (size_t)m * G4;
            const float g0 = EPS_F * gr[j];
            const float g1 = EPS_F * gr[HID + j];
            const float g2 = EPS_F * gr[2 * HID + j];
            const float g3 = EPS_F * gr[3 * HID + j];

            const float xn  = xnorm[t * BATCH + m];
            const float hn  = hnr[rr];
            const float nrm = sqrtf(xn + hn + 2.0f);
            const float inv = 1.0f / nrm;
            const float s   = (xn + hn) * inv;

            const float dpi = g0 * (s + inv) + g1 * inv + dhi;
            const float dpf = g1 * s + (g2 + g3) * inv + dhf;
            const float dpg = g2 * s + (g0 + g1) * inv + dhg;
            const float dpo = g3 * s + (g2 + g3) * inv + dho;

            const float i_  = sigmoidf_(pi);
            const float f_  = sigmoidf_(pf);
            const float gg  = tanhf(pg);
            const float o_  = sigmoidf_(po);

            const float di  = i_ * (1.0f - i_) * dpi;
            const float df  = f_ * (1.0f - f_) * dpf;
            const float dgg = (1.0f - gg * gg) * dpg;
            const float do_ = o_ * (1.0f - o_) * dpo;

            const size_t sidx = (size_t)m * HID + j;
            const float c  = c_st[sidx];
            const float dc = dc_st[sidx];

            const float c2  = f_ * c + i_ * gg;
            const float dc2 = df * c + f_ * dc + di * gg + i_ * dgg;
            const float tc  = tanhf(c2);
            const float h2  = o_ * tc;
            const float dh2 = do_ * tc + o_ * (1.0f - tc * tc) * dc2;

            c_st[sidx]    = c2;
            dc_st[sidx]   = dc2;
            hb_out[sidx]  = (bf16_t)h2;
            dhb_out[sidx] = (bf16_t)dh2;

            const size_t o0 = ((size_t)t * BATCH + m) * HID + j;
            out[o0] = h2;
            out[(size_t)T_STEPS * BATCH * HID + o0] = dh2;

            // stage h2^2 into the (already consumed) gate-0 pre slot:
            // this slot is read+written only by this thread -> no hazard.
            lds_pre[0 * 2048 + m * 32 + jj] = h2 * h2;
        }

        // --- Phase 5: convert x_{t+1} slice to bf16 (double buffer) --------
        if (t + 1 < T_STEPS) {
            const float* xsrc = x + (size_t)(t + 1) * BATCH * DIM;
            bf16_t* xdst = xb + (size_t)nxt * BATCH * DIM;
#pragma unroll
            for (int rr = 0; rr < 8; ++rr) {
                const int i = wg * 2048 + tid + rr * 256;
                xdst[i] = (bf16_t)xsrc[i];
            }
        }
        __syncthreads();

        // --- Phase 6: deterministic per-WG |h|^2 partial for next step -----
        if (tid < BATCH) {
            float s = 0.0f;
            for (int q = 0; q < 32; ++q) s += lds_pre[tid * 32 + q];
            hp_out[wg * BATCH + tid] = s;
        }

        // --- Phase 7: device-wide barrier ----------------------------------
        grid_barrier(bar_cnt, bar_gen, t);
    }
}

// ---------------------------------------------------------------------------
// Host launcher
// ---------------------------------------------------------------------------

extern "C" void kernel_launch(void* const* d_in, const int* in_sizes, int n_in,
                              void* d_out, int out_size, void* d_ws, size_t ws_size,
                              hipStream_t stream) {
    const float* x   = (const float*)d_in[0];   // [T,B,D]
    const float* g   = (const float*)d_in[1];   // [T,B,4H]
    const float* Wih = (const float*)d_in[2];   // [4H,D]
    const float* Whh = (const float*)d_in[3];   // [4H,H]
    const float* bih = (const float*)d_in[4];   // [4H]
    const float* bhh = (const float*)d_in[5];   // [4H]
    float* out = (float*)d_out;                 // [2,T,B,H]

    // Workspace carving (256B aligned slots)
    char* p = (char*)d_ws;
    auto carve = [&p](size_t bytes) -> void* {
        void* r = (void*)p;
        p += (bytes + 255) & ~(size_t)255;
        return r;
    };
    bf16_t* Wbih  = (bf16_t*)carve((size_t)G4 * DIM * sizeof(bf16_t));
    bf16_t* Wbhh  = (bf16_t*)carve((size_t)G4 * HID * sizeof(bf16_t));
    float*  bsum  = (float*)carve((size_t)G4 * sizeof(float));
    float*  xnorm = (float*)carve((size_t)T_STEPS * BATCH * sizeof(float));
    // zero-initialized region (contiguous)
    char* zbase = p;
    unsigned int* bar_cnt = (unsigned int*)carve(sizeof(unsigned int));
    unsigned int* bar_gen = (unsigned int*)carve(sizeof(unsigned int));
    float*  hpart = (float*)carve((size_t)2 * NWG * BATCH * sizeof(float));
    float*  c_st  = (float*)carve((size_t)BATCH * HID * sizeof(float));
    float*  dc_st = (float*)carve((size_t)BATCH * HID * sizeof(float));
    bf16_t* hb    = (bf16_t*)carve((size_t)2 * BATCH * HID * sizeof(bf16_t));
    bf16_t* dhb   = (bf16_t*)carve((size_t)2 * BATCH * HID * sizeof(bf16_t));
    bf16_t* xb    = (bf16_t*)carve((size_t)2 * BATCH * DIM * sizeof(bf16_t));
    const size_t zbytes = (size_t)(p - zbase);

    // Prep: bf16 weights + fused bias, zero state, |x_t|^2 table, bf16 x_0
    prep_convert<<<(G4 * DIM) / 256, 256, 0, stream>>>(Wih, Whh, bih, bhh,
                                                       Wbih, Wbhh, bsum);
    const int zwords = (int)(zbytes / 4);
    prep_zero<<<(zwords + 255) / 256, 256, 0, stream>>>((uint32_t*)zbase, zwords);
    prep_xnorm<<<T_STEPS * BATCH, 256, 0, stream>>>(x, xnorm);
    prep_x0<<<(BATCH * DIM) / 256, 256, 0, stream>>>(x, xb);

    // One persistent kernel runs all T timesteps with device-wide barriers.
    lstm_persist<<<NWG, 256, 0, stream>>>(x, g, Wbih, Wbhh, bsum, xnorm,
                                          hpart, xb, hb, dhb, c_st, dc_st,
                                          out, bar_cnt, bar_gen);
}
